// DynamicConv_77051713290507
// MI455X (gfx1250) — matrix operand order repaired and checked by reading, not compiled, not run on hardware
//
#include <hip/hip_runtime.h>
#include <hip/hip_bf16.h>

// MI455X / gfx1250, wave32. All heavy contractions on v_wmma_f32_16x16x32_f16.

typedef __attribute__((ext_vector_type(16))) _Float16 v16h;
typedef __attribute__((ext_vector_type(8)))  float    v8f;

#define BB 8
#define NN 300
#define BN (BB * NN)            // 2400
#define IN_DIM 256
#define OUT_DIM 256
#define GG 4
#define PTS 32                  // in_points == out_points
#define CGD 64                  // CG == DG
#define MPAR (CGD * CGD)        // 4096 per group
#define SPAR (PTS * PTS)        // 1024 per group
#define FLAT (GG * PTS * CGD)   // 8192
#define TEMPER 8.0f             // sqrt(64)

static __device__ __forceinline__ v8f wmma_f16(v16h a, v16h b, v8f c) {
  // D = A(16x32 f16) * B(32x16 f16) + C(16x16 f32)
  return __builtin_amdgcn_wmma_f32_16x16x32_f16(false, a, false, b, (short)0, c, false, false);
}

// ---- fragment loaders (wave32; contiguous k = 16*(lane/16) + e, same order in A and B) ----
// A 16x32 f16: lane L(<16) row=L, K in {0..15}; lane L+16 same row, K in {16..31}
static __device__ __forceinline__ v16h frag_A(const _Float16* src, int ld) {
  int l = threadIdx.x & 31;
  int row = l & 15;
  int kb = (l >> 4) << 4;
  v16h a;
#pragma unroll
  for (int e = 0; e < 16; ++e) {
    int k = kb + e;
    a[e] = src[row * ld + k];
  }
  return a;
}

static __device__ __forceinline__ v16h frag_A_f32(const float* src, int ld) {
  int l = threadIdx.x & 31;
  int row = l & 15;
  int kb = (l >> 4) << 4;
  v16h a;
#pragma unroll
  for (int e = 0; e < 16; ++e) {
    int k = kb + e;
    a[e] = (_Float16)src[row * ld + k];
  }
  return a;
}

// B 32x16 f16, memory row-major K x N: element (k,n) = src[k*ld + n]
static __device__ __forceinline__ v16h frag_B(const _Float16* src, int ld) {
  int l = threadIdx.x & 31;
  int n = l & 15;
  int kb = (l >> 4) << 4;
  v16h b;
#pragma unroll
  for (int e = 0; e < 16; ++e) b[e] = src[(kb + e) * ld + n];
  return b;
}

// B where memory is N x K row-major (contiguous in k): element (k,n) = src[n*ld + k]
static __device__ __forceinline__ v16h frag_B_T(const _Float16* src, int ld) {
  int l = threadIdx.x & 31;
  const _Float16* p = src + (size_t)(l & 15) * ld + ((l >> 4) << 4);
  v16h b;
#pragma unroll
  for (int e = 0; e < 16; ++e) b[e] = p[e];
  return b;
}

// ---------------- phase 0: f32 -> f16 conversion ----------------
__global__ void cvt_f32_f16(const float* __restrict__ src, _Float16* __restrict__ dst, int n) {
  int i = blockIdx.x * blockDim.x + threadIdx.x;
  if (i < n) dst[i] = (_Float16)src[i];
}

// ---------------- phase 1: generator GEMMs ----------------
// Y[bn,col] = sum_k X[bn,k] * W[col,k] + bias[col]   (Y stored f16)
// tr64: store transposed within each 4096-col group block (c*64+d -> d*64+c),
//       so downstream B-fragments read contiguous-k (vectorizes to b128 loads).
// grid: (BN/16, OUT/128); block 256 threads = 8 waves, one 16x16 n-tile each.
__global__ __launch_bounds__(256) void gen_gemm(const _Float16* __restrict__ Xh,
                                                const _Float16* __restrict__ Wh,
                                                const float* __restrict__ bias,
                                                _Float16* __restrict__ Y, int OUT, int tr64) {
  __shared__ _Float16 As[16 * IN_DIM];
  int bn0 = blockIdx.x * 16;
  for (int i = threadIdx.x; i < 16 * IN_DIM; i += 256) {
    int r = i >> 8, c = i & 255;
    As[i] = Xh[(size_t)(bn0 + r) * IN_DIM + c];
  }
  __syncthreads();
  int w = threadIdx.x >> 5;
  int col0 = blockIdx.y * 128 + w * 16;
  v8f acc = {};
#pragma unroll
  for (int k0 = 0; k0 < IN_DIM; k0 += 32) {
    v16h a = frag_A(As + k0, IN_DIM);
    v16h b = frag_B_T(Wh + (size_t)col0 * IN_DIM + k0, IN_DIM);
    acc = wmma_f16(a, b, acc);
  }
  int l = threadIdx.x & 31;
  int n = l & 15;
  int mo = (l >> 4) << 3;
  int col = col0 + n;
  float bia = bias[col];
  int scol = tr64 ? ((col & ~4095) | ((col & 63) << 6) | ((col >> 6) & 63)) : col;
#pragma unroll
  for (int r = 0; r < 8; ++r)
    Y[(size_t)(bn0 + mo + r) * OUT + scol] = (_Float16)(acc[r] + bia);
}

// ---------------- phase 2: fused per-(bn, g) ----------------
// LN over full 32x64 tile (2048 elems) via LDS float atomics, then ReLU.
static __device__ __forceinline__ void ln_relu_store(v8f c0, v8f c1, _Float16* dst, int ld,
                                                     int w, float* red) {
  if (threadIdx.x == 0) { red[0] = 0.f; red[1] = 0.f; }
  __syncthreads();
  float s = 0.f, ss = 0.f;
#pragma unroll
  for (int r = 0; r < 8; ++r) {
    s += c0[r] + c1[r];
    ss += c0[r] * c0[r] + c1[r] * c1[r];
  }
  atomicAdd(&red[0], s);
  atomicAdd(&red[1], ss);
  __syncthreads();
  const float invN = 1.0f / 2048.0f;
  float mean = red[0] * invN;
  float var = red[1] * invN - mean * mean;
  float rstd = rsqrtf(var + 1e-5f);
  int l = threadIdx.x & 31;
  int n = (l & 15) + w * 16;
  int mo = (l >> 4) << 3;
#pragma unroll
  for (int r = 0; r < 8; ++r) {
    dst[(size_t)(mo + r) * ld + n]      = (_Float16)fmaxf((c0[r] - mean) * rstd, 0.f);
    dst[(size_t)(16 + mo + r) * ld + n] = (_Float16)fmaxf((c1[r] - mean) * rstd, 0.f);
  }
  __syncthreads();
}

__global__ __launch_bounds__(128)
void fused_group(const float* __restrict__ feats,
                 const _Float16* __restrict__ Mws, const _Float16* __restrict__ Sws,
                 const _Float16* __restrict__ Qws, const _Float16* __restrict__ Vws,
                 const float* __restrict__ k_w, const float* __restrict__ k_b,
                 _Float16* __restrict__ FMS, _Float16* __restrict__ FC) {
  __shared__ _Float16 f_s[PTS * CGD];
  __shared__ _Float16 fm_s[PTS * CGD];
  __shared__ _Float16 v_s[PTS * CGD];
  __shared__ _Float16 k_s[PTS * PTS];
  __shared__ _Float16 sc_s[PTS * PTS];
  __shared__ float sc_f[PTS * PTS];
  __shared__ float red[2];

  int bn = blockIdx.x;
  int g = blockIdx.y;
  int w = threadIdx.x >> 5;
  int l = threadIdx.x & 31;
  int nit = l & 15;
  int mo = (l >> 4) << 3;

  const float* fp = feats + (size_t)(bn * GG + g) * (PTS * CGD);
  for (int i = threadIdx.x; i < PTS * CGD; i += 128) f_s[i] = (_Float16)fp[i];
  __syncthreads();

  // M/V group blocks are stored TRANSPOSED (d-major): element (c,d) at Mg[d*64+c]
  const _Float16* Mg = Mws + (size_t)bn * (GG * MPAR) + (size_t)g * MPAR;
  const _Float16* Vg = Vws + (size_t)bn * (GG * MPAR) + (size_t)g * MPAR;
  const _Float16* Sg = Sws + (size_t)bn * (GG * SPAR) + (size_t)g * SPAR;
  const _Float16* Qg = Qws + (size_t)bn * (GG * SPAR) + (size_t)g * SPAR;

  // fM = relu(LN2d(f @ M));  output 32x64, wave w owns n-cols [16w,16w+16)
  v8f a0 = {}, a1 = {};
#pragma unroll
  for (int k0 = 0; k0 < CGD; k0 += 32) {
    v16h fa0 = frag_A(f_s + k0, CGD);
    v16h fa1 = frag_A(f_s + 16 * CGD + k0, CGD);
    v16h mb  = frag_B_T(Mg + (size_t)(w * 16) * CGD + k0, CGD);  // contiguous in k=c
    a0 = wmma_f16(fa0, mb, a0);
    a1 = wmma_f16(fa1, mb, a1);
  }
  ln_relu_store(a0, a1, fm_s, CGD, w, red);

  // fMS = relu(LN2d(S @ fM)) -> global (layout g*2048 + o*64 + d)
  {
    v8f z0 = {}, z1 = {};
    v16h sa0 = frag_A(Sg, PTS);
    v16h sa1 = frag_A(Sg + 16 * PTS, PTS);
    v16h fb  = frag_B(fm_s + w * 16, CGD);
    z0 = wmma_f16(sa0, fb, z0);
    z1 = wmma_f16(sa1, fb, z1);
    ln_relu_store(z0, z1, FMS + (size_t)bn * FLAT + (size_t)g * (PTS * CGD), CGD, w, red);
  }

  // v = relu(LN2d(f @ Vw))
  a0 = (v8f){}; a1 = (v8f){};
#pragma unroll
  for (int k0 = 0; k0 < CGD; k0 += 32) {
    v16h fa0 = frag_A(f_s + k0, CGD);
    v16h fa1 = frag_A(f_s + 16 * CGD + k0, CGD);
    v16h vb  = frag_B_T(Vg + (size_t)(w * 16) * CGD + k0, CGD);  // contiguous in k=c
    a0 = wmma_f16(fa0, vb, a0);
    a1 = wmma_f16(fa1, vb, a1);
  }
  ln_relu_store(a0, a1, v_s, CGD, w, red);

  // k[i,p] = sum_d kw[g,i,d] * v[p,d] + kb[g,i]   (32x32, 2x2 tiles, 1 per wave)
  {
    int mt = w >> 1, nt = w & 1;
    v8f kc = {};
#pragma unroll
    for (int k0 = 0; k0 < CGD; k0 += 32) {
      v16h ka = frag_A_f32(k_w + (size_t)(g * PTS + mt * 16) * CGD + k0, CGD);
      v16h vb = frag_B_T(v_s + (size_t)(nt * 16) * CGD + k0, CGD);  // (k=d,n=p)=v_s[p*64+d]
      kc = wmma_f16(ka, vb, kc);
    }
#pragma unroll
    for (int r = 0; r < 8; ++r) {
      int i = mt * 16 + mo + r;
      int p = nt * 16 + nit;
      k_s[i * PTS + p] = (_Float16)(kc[r] + k_b[g * PTS + i]);
    }
  }
  __syncthreads();

  // scores = (q @ k) / TEMPER, row-softmax over p
  {
    int mt = w >> 1, nt = w & 1;
    v8f sc = {};
    v16h qa = frag_A(Qg + (size_t)(mt * 16) * PTS, PTS);
    v16h kb2 = frag_B(k_s + nt * 16, PTS);
    sc = wmma_f16(qa, kb2, sc);
#pragma unroll
    for (int r = 0; r < 8; ++r)
      sc_f[(mt * 16 + mo + r) * PTS + nt * 16 + nit] = sc[r] * (1.0f / TEMPER);
  }
  __syncthreads();
  if (threadIdx.x < PTS) {
    int o = threadIdx.x;
    float mx = -1e30f;
    for (int p = 0; p < PTS; ++p) mx = fmaxf(mx, sc_f[o * PTS + p]);
    float sum = 0.f;
    for (int p = 0; p < PTS; ++p) sum += __expf(sc_f[o * PTS + p] - mx);
    float inv = 1.0f / sum;
    for (int p = 0; p < PTS; ++p)
      sc_s[o * PTS + p] = (_Float16)(__expf(sc_f[o * PTS + p] - mx) * inv);
  }
  __syncthreads();

  // fc = relu(LN2d(sc @ v)) -> global
  {
    v8f z0 = {}, z1 = {};
    v16h sa0 = frag_A(sc_s, PTS);
    v16h sa1 = frag_A(sc_s + 16 * PTS, PTS);
    v16h vb  = frag_B(v_s + w * 16, CGD);
    z0 = wmma_f16(sa0, vb, z0);
    z1 = wmma_f16(sa1, vb, z1);
    ln_relu_store(z0, z1, FC + (size_t)bn * FLAT + (size_t)g * (PTS * CGD), CGD, w, red);
  }
}

// ---------------- phase 3: out projection + residual LayerNorm ----------------
// out[bn,:] = LN(qv[bn,:] + X[bn,:] @ W^T + bias) * lnw + lnb
// grid: BN/16 blocks, 256 threads (8 waves; wave w owns cols [32w, 32w+32))
__global__ __launch_bounds__(256)
void proj_ln(const _Float16* __restrict__ X, const _Float16* __restrict__ Wh,
             const float* __restrict__ bias, const float* __restrict__ qv,
             const float* __restrict__ lnw, const float* __restrict__ lnb,
             float* __restrict__ out) {
  __shared__ _Float16 As[16 * 256];
  __shared__ float os[16 * 256];
  int bn0 = blockIdx.x * 16;
  int w = threadIdx.x >> 5;
  int l = threadIdx.x & 31;
  v8f acc0 = {}, acc1 = {};
  for (int kc = 0; kc < FLAT; kc += 256) {
    for (int i = threadIdx.x; i < 16 * 256; i += 256) {
      int r = i >> 8, c = i & 255;
      As[i] = X[(size_t)(bn0 + r) * FLAT + kc + c];
    }
    __syncthreads();
#pragma unroll
    for (int k0 = 0; k0 < 256; k0 += 32) {
      v16h a = frag_A(As + k0, 256);
      v16h b0 = frag_B_T(Wh + (size_t)(w * 32) * FLAT + kc + k0, FLAT);
      v16h b1 = frag_B_T(Wh + (size_t)(w * 32 + 16) * FLAT + kc + k0, FLAT);
      acc0 = wmma_f16(a, b0, acc0);
      acc1 = wmma_f16(a, b1, acc1);
    }
    __syncthreads();
  }
  int n = l & 15;
  int mo = (l >> 4) << 3;
  int col0 = w * 32;
  float b0f = bias[col0 + n];
  float b1f = bias[col0 + 16 + n];
#pragma unroll
  for (int r = 0; r < 8; ++r) {
    os[(mo + r) * 256 + col0 + n]      = acc0[r] + b0f;
    os[(mo + r) * 256 + col0 + 16 + n] = acc1[r] + b1f;
  }
  __syncthreads();
  if (threadIdx.x < 16) {
    int row = threadIdx.x;
    const float* q = qv + (size_t)(bn0 + row) * 256;
    float s = 0.f, ss = 0.f;
    for (int c = 0; c < 256; ++c) {
      float x = q[c] + os[row * 256 + c];
      os[row * 256 + c] = x;
      s += x; ss += x * x;
    }
    float mean = s * (1.0f / 256.0f);
    float rstd = rsqrtf(ss * (1.0f / 256.0f) - mean * mean + 1e-5f);
    float* o = out + (size_t)(bn0 + row) * 256;
    for (int c = 0; c < 256; ++c)
      o[c] = (os[row * 256 + c] - mean) * rstd * lnw[c] + lnb[c];
  }
}

extern "C" void kernel_launch(void* const* d_in, const int* in_sizes, int n_in,
                              void* d_out, int out_size, void* d_ws, size_t ws_size,
                              hipStream_t stream) {
  const float* feats = (const float*)d_in[0];
  const float* qv    = (const float*)d_in[1];
  // d_in[2..6] (sample points / offsets / boxes) are unused by the reference outputs.
  const float* m_w = (const float*)d_in[7];   const float* m_b = (const float*)d_in[8];
  const float* s_w = (const float*)d_in[9];   const float* s_b = (const float*)d_in[10];
  const float* q_w = (const float*)d_in[11];  const float* q_b = (const float*)d_in[12];
  const float* v_w = (const float*)d_in[13];  const float* v_b = (const float*)d_in[14];
  const float* k_w = (const float*)d_in[15];  const float* k_b = (const float*)d_in[16];
  const float* Wv_w = (const float*)d_in[17]; const float* Wv_b = (const float*)d_in[18];
  const float* Wv2_w = (const float*)d_in[19];const float* Wv2_b = (const float*)d_in[20];
  const float* lnA_w = (const float*)d_in[21];const float* lnA_b = (const float*)d_in[22];
  const float* lnB_w = (const float*)d_in[23];const float* lnB_b = (const float*)d_in[24];
  float* out = (float*)d_out;

  char* ws = (char*)d_ws;
  size_t off = 0;
  auto alloc_h = [&](size_t elems) { _Float16* p = (_Float16*)(ws + off); off += elems * sizeof(_Float16); return p; };

  _Float16* qv_h  = alloc_h((size_t)BN * IN_DIM);
  _Float16* mw_h  = alloc_h((size_t)GG * MPAR * IN_DIM);
  _Float16* sw_h  = alloc_h((size_t)GG * SPAR * IN_DIM);
  _Float16* qw_h  = alloc_h((size_t)GG * SPAR * IN_DIM);
  _Float16* vw_h  = alloc_h((size_t)GG * MPAR * IN_DIM);
  _Float16* wv_h  = alloc_h((size_t)OUT_DIM * FLAT);
  _Float16* wv2_h = alloc_h((size_t)OUT_DIM * FLAT);
  _Float16* Mws   = alloc_h((size_t)BN * GG * MPAR);
  _Float16* Sws   = alloc_h((size_t)BN * GG * SPAR);
  _Float16* Qws   = alloc_h((size_t)BN * GG * SPAR);
  _Float16* Vws   = alloc_h((size_t)BN * GG * MPAR);
  _Float16* FMS   = alloc_h((size_t)BN * FLAT);
  _Float16* FC    = alloc_h((size_t)BN * FLAT);
  (void)ws_size; (void)in_sizes; (void)n_in; (void)out_size;

  auto cvt = [&](const float* s, _Float16* d, int n) {
    cvt_f32_f16<<<(n + 255) / 256, 256, 0, stream>>>(s, d, n);
  };
  cvt(qv,   qv_h,  BN * IN_DIM);
  cvt(m_w,  mw_h,  GG * MPAR * IN_DIM);
  cvt(s_w,  sw_h,  GG * SPAR * IN_DIM);
  cvt(q_w,  qw_h,  GG * SPAR * IN_DIM);
  cvt(v_w,  vw_h,  GG * MPAR * IN_DIM);
  cvt(Wv_w, wv_h,  OUT_DIM * FLAT);
  cvt(Wv2_w,wv2_h, OUT_DIM * FLAT);

  // generator GEMMs (M/V stored group-transposed for contiguous B-fragment reads)
  gen_gemm<<<dim3(BN / 16, (GG * MPAR) / 128), 256, 0, stream>>>(qv_h, mw_h, m_b, Mws, GG * MPAR, 1);
  gen_gemm<<<dim3(BN / 16, (GG * SPAR) / 128), 256, 0, stream>>>(qv_h, sw_h, s_b, Sws, GG * SPAR, 0);
  gen_gemm<<<dim3(BN / 16, (GG * SPAR) / 128), 256, 0, stream>>>(qv_h, qw_h, q_b, Qws, GG * SPAR, 0);
  gen_gemm<<<dim3(BN / 16, (GG * MPAR) / 128), 256, 0, stream>>>(qv_h, vw_h, v_b, Vws, GG * MPAR, 1);

  // fused per-(bn, g) branch math
  fused_group<<<dim3(BN, GG), 128, 0, stream>>>(feats, Mws, Sws, Qws, Vws, k_w, k_b, FMS, FC);

  // output projections + residual LN: reg then cls halves of d_out
  proj_ln<<<BN / 16, 256, 0, stream>>>(FMS, wv_h, Wv_b, qv, lnA_w, lnA_b, out);
  proj_ln<<<BN / 16, 256, 0, stream>>>(FC, wv2_h, Wv2_b, qv, lnB_w, lnB_b, out + (size_t)BN * OUT_DIM);
}
